// CustomGRU_81535659147985
// MI455X (gfx1250) — matrix operand order repaired and checked by reading, not compiled, not run on hardware
//
#include <hip/hip_runtime.h>
#include <cstdint>
#include <cstddef>
#include <utility>

// ---------------------------------------------------------------------------
// CDNA5 (gfx1250) bidirectional 2-layer GRU using v_wmma_f32_16x16x32_bf16.
// Round 2: software-pipelined K loops (register double buffering) and a bf16
// mirror of the recurrent h state so A-fragments load as b128 with no cvt.
// ---------------------------------------------------------------------------

typedef __attribute__((ext_vector_type(16))) __bf16 v16bf;
typedef __attribute__((ext_vector_type(8)))  float  v8f;

static __device__ __forceinline__ v8f wmma_bf16(v16bf a, v16bf b, v8f c) {
  // 8 args: (neg_a, A, neg_b, B, c_mod, C, reuse_a, reuse_b)
  return __builtin_amdgcn_wmma_f32_16x16x32_bf16(false, a, false, b, (short)0, c,
                                                 false, false);
}

// A fragment: 16x32 (MxK) bf16, row-major source, p0 -> (row0, k0), ld = leading dim.
// Layout used here (contiguous k, same as load_Bt, not the ISA table order): lane%16 = row M; lanes<16 hold K {0..15}, lanes>=16 hold
// K {16..31}; vgpr i carries 2 consecutive K values.
static __device__ __forceinline__ v16bf load_A_bf16(const __bf16* p0, int ld, int lane) {
  int r = lane & 15;
  int half = lane >> 4;
  const __bf16* p = p0 + (size_t)r * ld + half * 16;
  v16bf a;
#pragma unroll
  for (int i = 0; i < 8; ++i) {
    int kb = i << 1;   // i*2
    a[2 * i]     = p[kb];
    a[2 * i + 1] = p[kb + 1];
  }
  return a;
}

// B fragment: logical B(k,n) = W[n, k] (i.e. B = W^T), W row-major [Ntotal x K],
// ldw = K. Lane%16 = column N; lanes<16 cover K 0..15, lanes>=16 cover K 16..31.
static __device__ __forceinline__ v16bf load_Bt(const __bf16* W, int ldw, int n0,
                                                int k0, int lane) {
  int n = n0 + (lane & 15);
  const __bf16* p = W + (size_t)n * ldw + k0 + ((lane >> 4) << 4);
  v16bf b;
#pragma unroll
  for (int i = 0; i < 8; ++i) {
    b[2 * i]     = p[2 * i];
    b[2 * i + 1] = p[2 * i + 1];
  }
  return b;
}

// ---------------------------------------------------------------------------
// out[M x N] = Xbf16[M x K] @ W^T + bias   (W row-major [N x K], out f32)
// Block = 128 threads (4 waves); each wave: 16 rows x 64 cols (A reused 4x).
// K-loop register double-buffered: loads for k+32 issue before WMMAs for k.
// grid = (N/64, M/64)
// ---------------------------------------------------------------------------
__global__ __launch_bounds__(128) void gemm_bias_wmma(
    const __bf16* __restrict__ X, const __bf16* __restrict__ W,
    const float* __restrict__ bias, float* __restrict__ out, int K, int N) {
  int lane = threadIdx.x & 31;
  int wave = threadIdx.x >> 5;
  int row0 = blockIdx.y * 64 + wave * 16;
  int col0 = blockIdx.x * 64;

  const __bf16* xr = X + (size_t)row0 * K;

  v8f acc[4] = {};
  // prologue: fragments for k0 = 0
  v16bf a = load_A_bf16(xr, K, lane);
  v16bf b0 = load_Bt(W, K, col0 +  0, 0, lane);
  v16bf b1 = load_Bt(W, K, col0 + 16, 0, lane);
  v16bf b2 = load_Bt(W, K, col0 + 32, 0, lane);
  v16bf b3 = load_Bt(W, K, col0 + 48, 0, lane);

  for (int k0 = 32; k0 < K; k0 += 32) {
    // issue next-iteration loads before consuming current fragments
    v16bf an  = load_A_bf16(xr + k0, K, lane);
    v16bf b0n = load_Bt(W, K, col0 +  0, k0, lane);
    v16bf b1n = load_Bt(W, K, col0 + 16, k0, lane);
    v16bf b2n = load_Bt(W, K, col0 + 32, k0, lane);
    v16bf b3n = load_Bt(W, K, col0 + 48, k0, lane);
    acc[0] = wmma_bf16(a, b0, acc[0]);
    acc[1] = wmma_bf16(a, b1, acc[1]);
    acc[2] = wmma_bf16(a, b2, acc[2]);
    acc[3] = wmma_bf16(a, b3, acc[3]);
    a = an; b0 = b0n; b1 = b1n; b2 = b2n; b3 = b3n;
  }
  acc[0] = wmma_bf16(a, b0, acc[0]);
  acc[1] = wmma_bf16(a, b1, acc[1]);
  acc[2] = wmma_bf16(a, b2, acc[2]);
  acc[3] = wmma_bf16(a, b3, acc[3]);

  int rbase = row0 + ((lane >> 4) << 3);
  int cbase = col0 + (lane & 15);
#pragma unroll
  for (int j = 0; j < 4; ++j) {
    int col = cbase + j * 16;
    float bv = bias[col];
#pragma unroll
    for (int i = 0; i < 8; ++i)
      out[(size_t)(rbase + i) * N + col] = acc[j][i] + bv;
  }
}

// ---------------------------------------------------------------------------
// One GRU timestep for BOTH directions of a layer (blockIdx.y = dir).
// 128 threads = 4 waves; wave w owns rows 16w..16w+15 (B=64); block owns
// hidden columns [j0, j0+16). gh r/z/n tiles via WMMA over K=H from the bf16
// mirror of h; fused sigmoid/tanh epilogue writes h_next (f32 + bf16 mirror)
// and the y time slice. grid = (H/16, 2); t-dependence via stream ordering.
// ---------------------------------------------------------------------------
__global__ __launch_bounds__(128) void gru_step(
    const float*  __restrict__ hinf0,  const float*  __restrict__ hinf1,
    const __bf16* __restrict__ hinb0,  const __bf16* __restrict__ hinb1,
    float*        __restrict__ houtf0, float*        __restrict__ houtf1,
    __bf16*       __restrict__ houtb0, __bf16*       __restrict__ houtb1,
    const __bf16* __restrict__ Wh0, const __bf16* __restrict__ Wh1,
    const float*  __restrict__ bh0, const float*  __restrict__ bh1,
    const float*  __restrict__ xg0, const float*  __restrict__ xg1,
    float* __restrict__ y, int t) {
  const int H = 512, T = 256;
  int dir = blockIdx.y;
  const float*  hinf  = dir ? hinf1  : hinf0;
  const __bf16* hinb  = dir ? hinb1  : hinb0;
  float*        houtf = dir ? houtf1 : houtf0;
  __bf16*       houtb = dir ? houtb1 : houtb0;
  const __bf16* Wh = dir ? Wh1 : Wh0;
  const float*  bh = dir ? bh1 : bh0;
  const float*  xg = dir ? xg1 : xg0;
  int teff = dir ? (T - 1 - t) : t;

  int lane = threadIdx.x & 31;
  int wave = threadIdx.x >> 5;
  int row0 = wave * 16;
  int j0 = blockIdx.x * 16;

  const __bf16* ha = hinb + (size_t)row0 * H;

  v8f accR = {}, accZ = {}, accN = {};
  v16bf a  = load_A_bf16(ha, H, lane);
  v16bf bR = load_Bt(Wh, H, j0,         0, lane);
  v16bf bZ = load_Bt(Wh, H, H + j0,     0, lane);
  v16bf bN = load_Bt(Wh, H, 2 * H + j0, 0, lane);
#pragma unroll 4
  for (int k0 = 32; k0 < H; k0 += 32) {
    v16bf an  = load_A_bf16(ha + k0, H, lane);
    v16bf bRn = load_Bt(Wh, H, j0,         k0, lane);
    v16bf bZn = load_Bt(Wh, H, H + j0,     k0, lane);
    v16bf bNn = load_Bt(Wh, H, 2 * H + j0, k0, lane);
    accR = wmma_bf16(a, bR, accR);
    accZ = wmma_bf16(a, bZ, accZ);
    accN = wmma_bf16(a, bN, accN);
    a = an; bR = bRn; bZ = bZn; bN = bNn;
  }
  accR = wmma_bf16(a, bR, accR);
  accZ = wmma_bf16(a, bZ, accZ);
  accN = wmma_bf16(a, bN, accN);

  int row = row0 + ((lane >> 4) << 3);
  int col = j0 + (lane & 15);
  float bhr = bh[col], bhz = bh[H + col], bhn = bh[2 * H + col];
#pragma unroll
  for (int i = 0; i < 8; ++i) {
    int b = row + i;
    const float* xgp = xg + ((size_t)b * T + teff) * (3 * H);
    float gr = xgp[col]       + accR[i] + bhr;
    float gz = xgp[H + col]   + accZ[i] + bhz;
    float gn = xgp[2 * H + col];
    float r = 1.f / (1.f + __expf(-gr));
    float z = 1.f / (1.f + __expf(-gz));
    float n = tanhf(gn + r * (accN[i] + bhn));
    float hp = hinf[(size_t)b * H + col];
    float hn = (1.f - z) * n + z * hp;
    houtf[(size_t)b * H + col] = hn;
    houtb[(size_t)b * H + col] = (__bf16)hn;
    y[((size_t)b * T + teff) * (2 * H) + dir * H + col] = hn;
  }
}

// ---------------------------------------------------------------------------
// Utility kernels
// ---------------------------------------------------------------------------
__global__ void cvt_f32_bf16(const float* __restrict__ in, __bf16* __restrict__ out,
                             size_t n) {
  size_t i = (size_t)blockIdx.x * blockDim.x + threadIdx.x;
  size_t s = (size_t)gridDim.x * blockDim.x;
  for (; i < n; i += s) out[i] = (__bf16)in[i];
}

__global__ void zero_u32(uint32_t* __restrict__ p, size_t n) {
  size_t i = (size_t)blockIdx.x * blockDim.x + threadIdx.x;
  size_t s = (size_t)gridDim.x * blockDim.x;
  for (; i < n; i += s) p[i] = 0u;
}

__global__ void copy_f32(const float* __restrict__ src, float* __restrict__ dst,
                         size_t n) {
  size_t i = (size_t)blockIdx.x * blockDim.x + threadIdx.x;
  size_t s = (size_t)gridDim.x * blockDim.x;
  for (; i < n; i += s) dst[i] = src[i];
}

// ---------------------------------------------------------------------------
extern "C" void kernel_launch(void* const* d_in, const int* in_sizes, int n_in,
                              void* d_out, int out_size, void* d_ws, size_t ws_size,
                              hipStream_t stream) {
  (void)in_sizes; (void)n_in; (void)out_size; (void)ws_size;
  const int B = 64, T = 256, D = 512, H = 512;
  const int G = 3 * H;                 // 1536
  const size_t M0 = (size_t)B * T;     // 16384
  const size_t HS = (size_t)B * H;     // per-buffer h elements

  const float* x    = (const float*)d_in[0];
  const float* Wi[4] = {(const float*)d_in[1],  (const float*)d_in[5],
                        (const float*)d_in[9],  (const float*)d_in[13]};
  const float* Wh[4] = {(const float*)d_in[2],  (const float*)d_in[6],
                        (const float*)d_in[10], (const float*)d_in[14]};
  const float* bi[4] = {(const float*)d_in[3],  (const float*)d_in[7],
                        (const float*)d_in[11], (const float*)d_in[15]};
  const float* bh[4] = {(const float*)d_in[4],  (const float*)d_in[8],
                        (const float*)d_in[12], (const float*)d_in[16]};
  float* out = (float*)d_out;

  // ---- workspace layout ----
  char* ws = (char*)d_ws;
  size_t off = 0;
  auto alloc = [&](size_t bytes) -> char* {
    char* p = ws + off;
    off = (off + bytes + 255) & ~(size_t)255;
    return p;
  };
  __bf16* xbf   = (__bf16*)alloc(M0 * D * 2);
  float*  y0    = (float*) alloc(M0 * 2 * H * 4);
  __bf16* y0bf  = (__bf16*)alloc(M0 * 2 * H * 2);
  float*  xgF   = (float*) alloc(M0 * G * 4);
  float*  xgB   = (float*) alloc(M0 * G * 4);
  __bf16* wibf[4];
  wibf[0] = (__bf16*)alloc((size_t)G * D * 2);
  wibf[1] = (__bf16*)alloc((size_t)G * D * 2);
  wibf[2] = (__bf16*)alloc((size_t)G * 2 * H * 2);
  wibf[3] = (__bf16*)alloc((size_t)G * 2 * H * 2);
  __bf16* whbf[4];
  for (int i = 0; i < 4; ++i) whbf[i] = (__bf16*)alloc((size_t)G * H * 2);
  float*  hfb = (float*) alloc(8 * HS * 4);  // 8 f32 ping-pong h buffers
  __bf16* hbb = (__bf16*)alloc(8 * HS * 2);  // 8 bf16 mirrors
  float*  hf[8];
  __bf16* hb[8];
  for (int i = 0; i < 8; ++i) { hf[i] = hfb + (size_t)i * HS; hb[i] = hbb + (size_t)i * HS; }

  const int CB = 256, CG = 2048;

  // ---- precision conversion ----
  cvt_f32_bf16<<<CG, CB, 0, stream>>>(x, xbf, M0 * D);
  cvt_f32_bf16<<<CG, CB, 0, stream>>>(Wi[0], wibf[0], (size_t)G * D);
  cvt_f32_bf16<<<CG, CB, 0, stream>>>(Wi[1], wibf[1], (size_t)G * D);
  cvt_f32_bf16<<<CG, CB, 0, stream>>>(Wi[2], wibf[2], (size_t)G * 2 * H);
  cvt_f32_bf16<<<CG, CB, 0, stream>>>(Wi[3], wibf[3], (size_t)G * 2 * H);
  for (int i = 0; i < 4; ++i)
    cvt_f32_bf16<<<CG, CB, 0, stream>>>(Wh[i], whbf[i], (size_t)G * H);
  // zero all h state (f32 + bf16 mirrors) with one launch per region
  zero_u32<<<CG, CB, 0, stream>>>((uint32_t*)hfb, 8 * HS);          // f32
  zero_u32<<<CG, CB, 0, stream>>>((uint32_t*)hbb, 8 * HS / 2);      // bf16 pairs

  dim3 ggrid(G / 64, (unsigned)(M0 / 64));

  // ---- layer 0: input projections ----
  gemm_bias_wmma<<<ggrid, 128, 0, stream>>>(xbf, wibf[0], bi[0], xgF, D, G);
  gemm_bias_wmma<<<ggrid, 128, 0, stream>>>(xbf, wibf[1], bi[1], xgB, D, G);

  // ---- layer 0: recurrence (256 sequential step launches, ping-pong h) ----
  int iF = 0, oF = 1, iB = 2, oB = 3;
  for (int t = 0; t < T; ++t) {
    gru_step<<<dim3(H / 16, 2), 128, 0, stream>>>(
        hf[iF], hf[iB], hb[iF], hb[iB], hf[oF], hf[oB], hb[oF], hb[oB],
        whbf[0], whbf[1], bh[0], bh[1], xgF, xgB, y0, t);
    std::swap(iF, oF);
    std::swap(iB, oB);
  }
  // finals for layer 0 now in hf[iF] / hf[iB]
  float* finals = out + M0 * 2 * H;
  copy_f32<<<64, 256, 0, stream>>>(hf[iF], finals + 0 * HS, HS);
  copy_f32<<<64, 256, 0, stream>>>(hf[iB], finals + 1 * HS, HS);

  // ---- layer 1: input projections from layer-0 output ----
  cvt_f32_bf16<<<CG, CB, 0, stream>>>(y0, y0bf, M0 * 2 * H);
  gemm_bias_wmma<<<ggrid, 128, 0, stream>>>(y0bf, wibf[2], bi[2], xgF, 2 * H, G);
  gemm_bias_wmma<<<ggrid, 128, 0, stream>>>(y0bf, wibf[3], bi[3], xgB, 2 * H, G);

  // ---- layer 1: recurrence, writing y directly into d_out ----
  iF = 4; oF = 5; iB = 6; oB = 7;
  for (int t = 0; t < T; ++t) {
    gru_step<<<dim3(H / 16, 2), 128, 0, stream>>>(
        hf[iF], hf[iB], hb[iF], hb[iB], hf[oF], hf[oB], hb[oF], hb[oB],
        whbf[2], whbf[3], bh[2], bh[3], xgF, xgB, out, t);
    std::swap(iF, oF);
    std::swap(iB, oB);
  }
  copy_f32<<<64, 256, 0, stream>>>(hf[iF], finals + 2 * HS, HS);
  copy_f32<<<64, 256, 0, stream>>>(hf[iB], finals + 3 * HS, HS);
}
